// LSTMCellPeephole_63874753626449
// MI455X (gfx1250) — matrix operand order repaired and checked by reading, not compiled, not run on hardware
//
#include <hip/hip_runtime.h>

// ---------------------------------------------------------------------------
// Peephole LSTM cell, fused single kernel for MI455X (gfx1250, wave32).
//   gates = [x | hx] @ [W_ih | W_hh]^T  (+ biases), then elementwise epilogue.
// GEMM: M=4096 (batch), N=4096 (4 gates x H), K=2048.  68.7 GFLOP, ~96MB IO.
// Compute-bound -> bf16 hi/lo split-precision WMMA (3x v_wmma_f32_16x16x32_bf16
// per tile ~= fp32 accuracy at ~2.7x the native fp32-WMMA rate).
//
// Pipeline (v2): 2-stage software pipeline.
//  - global loads for stage ks+1 issued into registers BEFORE the WMMA work of
//    stage ks, so L2 latency hides under ~48 WMMAs/wave of XDL work;
//  - fp32 -> bf16 hi/lo conversion + ds_store land AFTER compute;
//  - double-buffered LDS (120KB of gfx1250's 320KB WGP LDS), ONE barrier/iter.
//
// Block: 256 thr = 8 waves; block tile = 128 batch x 64 hidden x 4 gates.
// Wave (r in 0..3, c in 0..1): 32x32 patch, all 4 gates in registers, so the
// sigmoid/tanh/peephole epilogue is fully fused (no gates buffer, no ws).
// ---------------------------------------------------------------------------

typedef __bf16 v4bf  __attribute__((ext_vector_type(4)));
typedef __bf16 v8bf  __attribute__((ext_vector_type(8)));
typedef __bf16 v16bf __attribute__((ext_vector_type(16)));
typedef float  v8f   __attribute__((ext_vector_type(8)));

#define SA 40  // padded LDS k-stride (32 + 8 elems = 80B) -> bank-spread, 16B aligned chunks

__device__ __forceinline__ v16bf join8(v8bf a, v8bf b) {
  return __builtin_shufflevector(a, b, 0, 1, 2, 3, 4, 5, 6, 7,
                                       8, 9, 10, 11, 12, 13, 14, 15);
}

__device__ __forceinline__ float sigmoid_fast(float x) {
  return 1.0f / (1.0f + __expf(-x));
}
__device__ __forceinline__ float tanh_fast(float x) {
  return 2.0f / (1.0f + __expf(-2.0f * x)) - 1.0f;
}

// Split one float4 into bf16 hi/lo planes and store 4 elems (8B) to LDS.
__device__ __forceinline__ void cvt_store4(__bf16* ph, __bf16* pl, const float4 f) {
  float fs[4] = {f.x, f.y, f.z, f.w};
  v4bf h, l;
#pragma unroll
  for (int i = 0; i < 4; ++i) {
    float v = fs[i];
    __bf16 hi = (__bf16)v;           // RNE fp32->bf16
    h[i] = hi;
    l[i] = (__bf16)(v - (float)hi);  // exact residual, then rounded
  }
  *reinterpret_cast<v4bf*>(ph) = h;
  *reinterpret_cast<v4bf*>(pl) = l;
}

__global__ __launch_bounds__(256) void lstm_peephole_wmma(
    const float* __restrict__ x,   const float* __restrict__ hx,
    const float* __restrict__ cx,
    const float* __restrict__ wih, const float* __restrict__ whh,
    const float* __restrict__ bih, const float* __restrict__ bhh,
    const float* __restrict__ wpi, const float* __restrict__ wpf,
    float* __restrict__ out) {
  constexpr int H = 1024;
  constexpr int BH = 4096 * 1024;  // hy size (cy follows)

  // Double-buffered staging tiles (bf16 hi/lo planes): 120KB total.
  __shared__ __bf16 Ah[2][128][SA];
  __shared__ __bf16 Al[2][128][SA];
  __shared__ __bf16 Wh[2][4][64][SA];
  __shared__ __bf16 Wl[2][4][64][SA];

  const int tid  = threadIdx.x;
  const int lane = tid & 31;
  const int ln   = lane & 15;       // N / M position inside 16-wide tile
  const int half = lane >> 4;       // lane-group selecting K sub-chunk
  const int r    = (tid >> 5) & 3;  // wave row  (batch)  0..3
  const int c    = (tid >> 7) & 1;  // wave col  (hidden) 0..1

  const int bm0 = blockIdx.x * 128;  // batch tile origin
  const int h0  = blockIdx.y * 64;   // hidden tile origin

  v8f acc[4][2][2];  // [gate][m-tile][n-tile], 16x16 f32 each (128 VGPRs)
#pragma unroll
  for (int g = 0; g < 4; ++g)
#pragma unroll
    for (int mi = 0; mi < 2; ++mi)
#pragma unroll
      for (int nj = 0; nj < 2; ++nj) {
        v8f z = {};
        acc[g][mi][nj] = z;
      }

  float4 ra[4];  // staged A data (4 float4 / thread)
  float4 rw[8];  // staged W data (8 float4 / thread)

  // Issue global loads for stage ks into ra/rw (no LDS side effects).
  auto load_stage = [&](int ks) {
    const int k0 = ks * 32;
    const float* Ab = (k0 < 1024) ? x : hx;
    const float* Wb = (k0 < 1024) ? wih : whh;
    const int kk = (k0 < 1024) ? k0 : (k0 - 1024);
#pragma unroll
    for (int i = 0; i < 4; ++i) {
      const int idx = tid + 256 * i;
      const int row = idx >> 3;
      const int kq  = (idx & 7) << 2;
      ra[i] = *reinterpret_cast<const float4*>(
          &Ab[(size_t)(bm0 + row) * 1024 + kk + kq]);
    }
#pragma unroll
    for (int i = 0; i < 8; ++i) {
      const int idx = tid + 256 * i;
      const int g   = idx >> 9;
      const int row = (idx >> 3) & 63;
      const int kq  = (idx & 7) << 2;
      rw[i] = *reinterpret_cast<const float4*>(
          &Wb[(size_t)(g * 1024 + h0 + row) * 1024 + kk + kq]);
    }
  };

  // Convert staged registers to bf16 hi/lo and store into LDS buffer p.
  auto store_stage = [&](int p) {
#pragma unroll
    for (int i = 0; i < 4; ++i) {
      const int idx = tid + 256 * i;
      const int row = idx >> 3;
      const int kq  = (idx & 7) << 2;
      cvt_store4(&Ah[p][row][kq], &Al[p][row][kq], ra[i]);
    }
#pragma unroll
    for (int i = 0; i < 8; ++i) {
      const int idx = tid + 256 * i;
      const int g   = idx >> 9;
      const int row = (idx >> 3) & 63;
      const int kq  = (idx & 7) << 2;
      cvt_store4(&Wh[p][g][row][kq], &Wl[p][g][row][kq], rw[i]);
    }
  };

  // Full WMMA work for one 32-wide K slab from LDS buffer p.
  auto compute = [&](int p) {
    // A fragments: lane = row ln,
    // contiguous 16 K at half*16 (same order as B).
    v16bf fah[2], fal[2];
#pragma unroll
    for (int mi = 0; mi < 2; ++mi) {
      const int row = 32 * r + 16 * mi + ln;
      const int kb  = half * 16;
      fah[mi] = join8(*reinterpret_cast<const v8bf*>(&Ah[p][row][kb]),
                      *reinterpret_cast<const v8bf*>(&Ah[p][row][kb + 8]));
      fal[mi] = join8(*reinterpret_cast<const v8bf*>(&Al[p][row][kb]),
                      *reinterpret_cast<const v8bf*>(&Al[p][row][kb + 8]));
    }
    // B fragments: lane = column ln, contiguous 16 K at half*16
    // (W row-major rows are exactly B columns -> no transpose).
#pragma unroll
    for (int g = 0; g < 4; ++g) {
#pragma unroll
      for (int nj = 0; nj < 2; ++nj) {
        const int col = 32 * c + 16 * nj + ln;
        const int kb  = half * 16;
        v16bf fbh = join8(*reinterpret_cast<const v8bf*>(&Wh[p][g][col][kb]),
                          *reinterpret_cast<const v8bf*>(&Wh[p][g][col][kb + 8]));
        v16bf fbl = join8(*reinterpret_cast<const v8bf*>(&Wl[p][g][col][kb]),
                          *reinterpret_cast<const v8bf*>(&Wl[p][g][col][kb + 8]));
#pragma unroll
        for (int mi = 0; mi < 2; ++mi) {
          v8f a0 = acc[g][mi][nj];
          // split-precision: Ah*Wh + Al*Wh + Ah*Wl  (Al*Wl term negligible)
          a0 = __builtin_amdgcn_wmma_f32_16x16x32_bf16(
              false, fah[mi], false, fbh, (short)0, a0, false, false);
          a0 = __builtin_amdgcn_wmma_f32_16x16x32_bf16(
              false, fal[mi], false, fbh, (short)0, a0, false, false);
          a0 = __builtin_amdgcn_wmma_f32_16x16x32_bf16(
              false, fah[mi], false, fbl, (short)0, a0, false, false);
          acc[g][mi][nj] = a0;
        }
      }
    }
  };

  // ---- 2-stage software pipeline over K = 2048 (64 slabs of 32) ----
  load_stage(0);
  store_stage(0);
  __syncthreads();

  for (int ks = 0; ks < 64; ++ks) {
    if (ks < 63) load_stage(ks + 1);  // loads in flight during compute
    compute(ks & 1);
    if (ks < 63) {
      store_stage((ks + 1) & 1);      // loadcnt wait lands here, after compute
      __syncthreads();
    }
  }

  // ---- fused epilogue: all 4 gates for each (b,h) live in this wave ----
  // C/D layout: VGPR v, lanes 0-15 -> M=v, lanes 16-31 -> M=v+8; N=ln.
#pragma unroll
  for (int nj = 0; nj < 2; ++nj) {
    const int h  = h0 + 32 * c + 16 * nj + ln;
    const float bi_ = bih[0 * 1024 + h] + bhh[0 * 1024 + h];
    const float bf_ = bih[1 * 1024 + h] + bhh[1 * 1024 + h];
    const float bc_ = bih[2 * 1024 + h] + bhh[2 * 1024 + h];
    const float bo_ = bih[3 * 1024 + h] + bhh[3 * 1024 + h];
    const float di  = wpi[(size_t)h * H + h];  // diagonal peephole
    const float df  = wpf[(size_t)h * H + h];  // reused for output gate (faithful)
#pragma unroll
    for (int mi = 0; mi < 2; ++mi) {
#pragma unroll
      for (int v = 0; v < 8; ++v) {
        const int b = bm0 + 32 * r + 16 * mi + 8 * half + v;
        const float cxv = cx[(size_t)b * H + h];
        float ig = acc[0][mi][nj][v] + bi_ + cxv * di;
        float fg = acc[1][mi][nj][v] + bf_ + cxv * df;
        float cg = acc[2][mi][nj][v] + bc_;
        float og = acc[3][mi][nj][v] + bo_;
        ig = sigmoid_fast(ig);
        fg = sigmoid_fast(fg);
        cg = tanh_fast(cg);
        const float cy = fg * cxv + ig * cg;
        og = sigmoid_fast(og + cy * df);
        const float hy = og * tanh_fast(cy);
        out[(size_t)b * H + h]      = hy;  // hy plane
        out[BH + (size_t)b * H + h] = cy;  // cy plane
      }
    }
  }
}

extern "C" void kernel_launch(void* const* d_in, const int* in_sizes, int n_in,
                              void* d_out, int out_size, void* d_ws, size_t ws_size,
                              hipStream_t stream) {
  const float* x   = (const float*)d_in[0];
  const float* hx  = (const float*)d_in[1];
  const float* cx  = (const float*)d_in[2];
  const float* wih = (const float*)d_in[3];
  const float* whh = (const float*)d_in[4];
  const float* bih = (const float*)d_in[5];
  const float* bhh = (const float*)d_in[6];
  const float* wpi = (const float*)d_in[7];
  const float* wpf = (const float*)d_in[8];
  // d_in[9] (W_peephole_o) is unused by the reference (it reuses W_peephole_f).
  float* out = (float*)d_out;

  dim3 grid(4096 / 128, 1024 / 64);  // 32 x 16 = 512 blocks
  lstm_peephole_wmma<<<grid, 256, 0, stream>>>(x, hx, cx, wih, whh, bih, bhh,
                                               wpi, wpf, out);
}
